// Convolve_91010357002742
// MI455X (gfx1250) — compile-verified, matrix operand order checked by reading, not run on hardware
//
#include <hip/hip_runtime.h>
#include <math.h>

// CDNA5 / gfx1250, wave32. All GEMMs via V_WMMA_F32_16X16X32_F16.
// Pipeline:
//   k_prep : E -> f16 copy; Wq/Wk/Wv/W1 -> pre-swizzled f16 B-fragments (global)
//   k_qproj: Q = lrelu(E @ Wq + bq)                       (WMMA, persistent)
//   k_attn : gather 16 nbrs, k/v = lrelu(nb@W+b) (WMMA), softmax+pool (VALU)
//   k_out  : H = lrelu([E|P] @ W1 + b1), row-L2-norm, BN  (WMMA, persistent)
// Weight fragments are staged global->LDS via the Tensor Data Mover
// (tensor_load_to_lds + s_wait_tensorcnt), with a plain copy as fallback.

typedef __attribute__((ext_vector_type(16))) _Float16 v16h;
typedef __attribute__((ext_vector_type(8)))  _Float16 v8h;
typedef __attribute__((ext_vector_type(8)))  float    v8f;
typedef __attribute__((ext_vector_type(4)))  float    v4f;
typedef __attribute__((ext_vector_type(4)))  unsigned int uint4v;
typedef __attribute__((ext_vector_type(8)))  unsigned int uint8v;

#define NN    50000
#define FF    128
#define HH    128
#define KNBR  16
#define TILES (NN / 16)        // 3125 -- NN % 16 == 0, every tile is full
#define ALPHA 0.3f
#define USE_TDM 1

__device__ __forceinline__ float lrelu(float x) { return x > 0.f ? x : ALPHA * x; }

// ---------------------------------------------------------------------------
// TDM: DMA a contiguous block (bytes, multiple of 8) from global into LDS.
// D# per CDNA5 ISA 8.3/8.4: group0 = {count=1, lds_addr, global_addr, type=2},
// group1 = {data_size=8B, tensor_dim0=units, tensor_dim1=1, tile_dim0=units,
//           tile_dim1=1, tensor_dim0_stride=units}. Issue from one wave;
// TENSORcnt tracks completion (EXEC is ignored by tensor instructions).
// ---------------------------------------------------------------------------
__device__ __forceinline__ void tdm_load_lds(const void* gsrc, unsigned lds_off,
                                             unsigned bytes) {
#if USE_TDM
  unsigned long long ga = (unsigned long long)gsrc;
  unsigned units = bytes >> 3;                       // 8-byte elements
  uint4v g0;
  g0[0] = 1u;                                        // count=1 (valid user D#)
  g0[1] = lds_off;                                   // lds_addr (bytes)
  g0[2] = (unsigned)(ga & 0xFFFFFFFFu);              // global_addr[31:0]
  g0[3] = (unsigned)((ga >> 32) & 0x01FFFFFFu) | (2u << 30);  // addr[56:32], type=2
  uint8v g1;
  g1[0] = 3u << 16;                                  // data_size = 8 bytes
  g1[1] = (units & 0xFFFFu) << 16;                   // tensor_dim0[15:0]
  g1[2] = ((units >> 16) & 0xFFFFu) | (1u << 16);    // tensor_dim0[31:16], dim1=1
  g1[3] = (units & 0xFFFFu) << 16;                   // tile_dim0
  g1[4] = 1u;                                        // tile_dim1 = 1
  g1[5] = units;                                     // tensor_dim0_stride[31:0]
  g1[6] = 0u;
  g1[7] = 0u;
  asm volatile("tensor_load_to_lds %0, %1" :: "s"(g0), "s"(g1) : "memory");
  __builtin_amdgcn_s_wait_tensorcnt(0);
#else
  (void)gsrc; (void)lds_off; (void)bytes;
#endif
}

// ---------------------------------------------------------------------------
// B-fragment index math (16-bit B operand of V_WMMA_F32_16X16X32_F16):
//   lanes 0-15 : col N = lane,    K = k0+{0..7, 16..23}
//   lanes16-31 : col N = lane-16, K = k0+{8..15, 24..31}
// Fragment fi = n0*KC + kc, stored lane-major, 16 halves (32B) per lane.
// ---------------------------------------------------------------------------
__device__ __forceinline__ void frag_convert(const float* __restrict__ W,
                                             _Float16* __restrict__ dst,
                                             int rows, int tid, int stride) {
  const int KC = rows >> 5;
  const int total = rows * HH;
  for (int i = tid; i < total; i += stride) {
    int elem = i & 15;
    int lane = (i >> 4) & 31;
    int fi   = i >> 9;
    int n0   = fi / KC;
    int kc   = fi - n0 * KC;
    int k    = kc * 32 + ((lane & 16) ? 8 : 0) + (elem < 8 ? elem : elem + 8);
    int col  = n0 * 16 + (lane & 15);
    dst[i] = (_Float16)W[k * HH + col];
  }
}

extern "C" __global__ void __launch_bounds__(256)
k_prep(const float* __restrict__ E,
       const float* __restrict__ Wq, const float* __restrict__ Wk,
       const float* __restrict__ Wv, const float* __restrict__ W1,
       _Float16* __restrict__ E16,
       _Float16* __restrict__ FWq, _Float16* __restrict__ FWk,
       _Float16* __restrict__ FWv, _Float16* __restrict__ FW1) {
  const int tid    = blockIdx.x * blockDim.x + threadIdx.x;
  const int stride = gridDim.x * blockDim.x;
  // embeddings -> f16, 4 elements per iteration (b128 in, b64 out)
  const int nv = (NN * FF) >> 2;
  for (int i = tid; i < nv; i += stride) {
    v4f x = ((const v4f*)E)[i];
    _Float16* p = E16 + i * 4;
    p[0] = (_Float16)x.x; p[1] = (_Float16)x.y;
    p[2] = (_Float16)x.z; p[3] = (_Float16)x.w;
  }
  frag_convert(Wq, FWq, FF, tid, stride);
  frag_convert(Wk, FWk, FF, tid, stride);
  frag_convert(Wv, FWv, FF, tid, stride);
  frag_convert(W1, FW1, FF + HH, tid, stride);
}

// straight copy of pre-swizzled fragments, global -> LDS (b128 / ds_store_b128)
__device__ __forceinline__ void stage_copy(const _Float16* __restrict__ src,
                                           _Float16* dst, int halves) {
  const v8h* s = (const v8h*)src;
  v8h* d = (v8h*)dst;
  const int n = halves >> 3;
  for (int i = threadIdx.x; i < n; i += blockDim.x) d[i] = s[i];
}

__device__ __forceinline__ v16h load_b_frag(const _Float16* lds, int KC,
                                            int n0, int kc, int lane) {
  const v16h* p = (const v16h*)lds;
  return p[(n0 * KC + kc) * 32 + lane];
}

// A operand (16x32 f16) from an f16 row pointer: two aligned 16B loads.
//   lanes 0-15 : row M = lane,    K = k0+{0..7, 16..23}
//   lanes16-31 : row M = lane-16, K = k0+{8..15, 24..31}
__device__ __forceinline__ v16h load_a_frag16(const _Float16* __restrict__ rowp,
                                              int k0, int lane) {
  const _Float16* p = rowp + k0 + ((lane & 16) ? 8 : 0);
  v8h lo = *(const v8h*)(p);
  v8h hi = *(const v8h*)(p + 16);
  return __builtin_shufflevector(lo, hi, 0, 1, 2, 3, 4, 5, 6, 7,
                                 8, 9, 10, 11, 12, 13, 14, 15);
}

// ---------------------------------------------------------------------------
// Kernel: Q = lrelu(E @ Wq + bq). Persistent; 16 rows per wave-tile.
// ---------------------------------------------------------------------------
extern "C" __global__ void __launch_bounds__(256)
k_qproj(const _Float16* __restrict__ E16, const _Float16* __restrict__ FWq,
        const float* __restrict__ bq, float* __restrict__ Q) {
  extern __shared__ _Float16 smem[];
  if (threadIdx.x == 0) tdm_load_lds(FWq, 0, FF * HH * 2);
  stage_copy(FWq, smem, FF * HH);
  __syncthreads();
  const int lane = threadIdx.x & 31;
  const int wave = threadIdx.x >> 5;
  const int step = gridDim.x * 8;
  const int hi   = (lane & 16) ? 8 : 0;

  float qb[8];
#pragma unroll
  for (int n0 = 0; n0 < 8; ++n0) qb[n0] = bq[n0 * 16 + (lane & 15)];

  for (int t = blockIdx.x * 8 + wave; t < TILES; t += step) {
    const int base = t * 16;
    const _Float16* rowp = E16 + (size_t)(base + (lane & 15)) * FF;
    {   // prefetch next tile's A rows into cache (global_prefetch_b8)
      int nt = t + step;
      if (nt < TILES)
        __builtin_prefetch(E16 + (size_t)(nt * 16 + (lane & 15)) * FF, 0, 0);
    }
    v16h a[4];
#pragma unroll
    for (int c = 0; c < 4; ++c) a[c] = load_a_frag16(rowp, c * 32, lane);

    // one address per tile; all 64 stores use constant immediate offsets
    float* outp = Q + (size_t)(base + hi) * HH + (lane & 15);
#pragma unroll
    for (int n0 = 0; n0 < 8; ++n0) {
      v8f acc = {};
#pragma unroll
      for (int c = 0; c < 4; ++c) {
        v16h b = load_b_frag(smem, 4, n0, c, lane);
        acc = __builtin_amdgcn_wmma_f32_16x16x32_f16(false, a[c], false, b,
                                                     (short)0, acc, false, false);
      }
#pragma unroll
      for (int r = 0; r < 8; ++r)
        outp[n0 * 16 + r * HH] = lrelu(acc[r] + qb[n0]);
    }
  }
}

// ---------------------------------------------------------------------------
// Kernel: per-node neighbor attention. One node per wave, persistent.
// ---------------------------------------------------------------------------
extern "C" __global__ void __launch_bounds__(256)
k_attn(const _Float16* __restrict__ E16, const int* __restrict__ nbr,
       const _Float16* __restrict__ FWk, const float* __restrict__ bk,
       const float* __restrict__ bv,
       const float* __restrict__ Q, _Float16* __restrict__ P16) {
  extern __shared__ _Float16 smem[];
  _Float16* swk = smem;                  // FWk and FWv are contiguous in ws:
  _Float16* swv = smem + FF * HH;        // stage both with one 64KB block
  if (threadIdx.x == 0) tdm_load_lds(FWk, 0, 2 * FF * HH * 2);
  stage_copy(FWk, swk, 2 * FF * HH);
  __syncthreads();
  const int lane = threadIdx.x & 31;
  const int wave = threadIdx.x >> 5;
  const int step = gridDim.x * 8;

  float kb[8], vb[8];
#pragma unroll
  for (int n0 = 0; n0 < 8; ++n0) {
    kb[n0] = bk[n0 * 16 + (lane & 15)];
    vb[n0] = bv[n0 * 16 + (lane & 15)];
  }

  for (int n = blockIdx.x * 8 + wave; n < NN; n += step) {
    {   // prefetch next node's neighbor list
      int nn = n + step;
      if (nn < NN) __builtin_prefetch(nbr + (size_t)nn * KNBR, 0, 0);
    }
    const int nb = nbr[(size_t)n * KNBR + (lane & 15)];
    const _Float16* arow = E16 + (size_t)nb * FF;
    v16h a[4];
#pragma unroll
    for (int c = 0; c < 4; ++c) a[c] = load_a_frag16(arow, c * 32, lane);

    v8f kf[8], vf[8];
#pragma unroll
    for (int n0 = 0; n0 < 8; ++n0) {
      v8f ak = {}, av = {};
#pragma unroll
      for (int c = 0; c < 4; ++c) {
        v16h bkf = load_b_frag(swk, 4, n0, c, lane);
        ak = __builtin_amdgcn_wmma_f32_16x16x32_f16(false, a[c], false, bkf,
                                                    (short)0, ak, false, false);
        v16h bvf = load_b_frag(swv, 4, n0, c, lane);
        av = __builtin_amdgcn_wmma_f32_16x16x32_f16(false, a[c], false, bvf,
                                                    (short)0, av, false, false);
      }
#pragma unroll
      for (int r = 0; r < 8; ++r) {
        kf[n0][r] = lrelu(ak[r] + kb[n0]);
        vf[n0][r] = lrelu(av[r] + vb[n0]);
      }
    }

    // scores[M] = sum_h q[h]*k[M,h]; lane's h per chunk = n0*16 + lane&15
    const float* qbase = Q + (size_t)n * HH + (lane & 15);
    float ql[8];
#pragma unroll
    for (int n0 = 0; n0 < 8; ++n0) ql[n0] = qbase[n0 * 16];

    float sc[8];
#pragma unroll
    for (int r = 0; r < 8; ++r) {
      float s = 0.f;
#pragma unroll
      for (int n0 = 0; n0 < 8; ++n0) s += ql[n0] * kf[n0][r];
#pragma unroll
      for (int m = 1; m <= 8; m <<= 1) s += __shfl_xor(s, m, 32);  // half-wave
      sc[r] = s;   // low half: score[r]; high half: score[8+r]
    }
    float so[8];
#pragma unroll
    for (int r = 0; r < 8; ++r) so[r] = __shfl_xor(sc[r], 16, 32);

    float mx = sc[0];
#pragma unroll
    for (int r = 0; r < 8; ++r) { mx = fmaxf(mx, sc[r]); mx = fmaxf(mx, so[r]); }
    float aw[8], denom = 0.f;
#pragma unroll
    for (int r = 0; r < 8; ++r) {
      aw[r] = __expf(sc[r] - mx);
      denom += aw[r] + __expf(so[r] - mx);
    }
    const float inv = 1.f / denom;
#pragma unroll
    for (int r = 0; r < 8; ++r) aw[r] *= inv;

    // pooled[h] = sum_M attn[M]*v[M,h]; lane^16 covers the other 8 rows
    _Float16* pbase = P16 + (size_t)n * HH + lane;
#pragma unroll
    for (int n0 = 0; n0 < 8; ++n0) {
      float p = 0.f;
#pragma unroll
      for (int r = 0; r < 8; ++r) p += aw[r] * vf[n0][r];
      p += __shfl_xor(p, 16, 32);
      if (lane < 16) pbase[n0 * 16] = (_Float16)p;
    }
  }
}

// ---------------------------------------------------------------------------
// Kernel: H = lrelu([E|P] @ W1 + b1); row L2-norm; BatchNorm affine.
// ---------------------------------------------------------------------------
extern "C" __global__ void __launch_bounds__(256)
k_out(const _Float16* __restrict__ E16, const _Float16* __restrict__ P16,
      const _Float16* __restrict__ FW1, const float* __restrict__ b1,
      const float* __restrict__ gamma, const float* __restrict__ beta,
      const float* __restrict__ mean, const float* __restrict__ var,
      float* __restrict__ out) {
  extern __shared__ _Float16 smem[];
  if (threadIdx.x == 0) tdm_load_lds(FW1, 0, (FF + HH) * HH * 2);
  stage_copy(FW1, smem, (FF + HH) * HH);
  __syncthreads();
  const int lane = threadIdx.x & 31;
  const int wave = threadIdx.x >> 5;
  const int step = gridDim.x * 8;
  const int hi   = (lane & 16) ? 8 : 0;

  float hb[8], g[8], bt[8], mn[8], iv[8];
#pragma unroll
  for (int n0 = 0; n0 < 8; ++n0) {
    const int col = n0 * 16 + (lane & 15);
    hb[n0] = b1[col];
    g[n0]  = gamma[col];
    bt[n0] = beta[col];
    mn[n0] = mean[col];
    iv[n0] = rsqrtf(var[col] + 1e-3f);
  }

  for (int t = blockIdx.x * 8 + wave; t < TILES; t += step) {
    const int base = t * 16;
    const int arow = base + (lane & 15);
    {
      int nt = t + step;
      if (nt < TILES)
        __builtin_prefetch(E16 + (size_t)(nt * 16 + (lane & 15)) * FF, 0, 0);
    }
    v8f acc[8] = {};
#pragma unroll
    for (int c = 0; c < 8; ++c) {   // K = 256: E16 rows then P16 rows
      const _Float16* rowp = (c < 4) ? (E16 + (size_t)arow * FF)
                                     : (P16 + (size_t)arow * HH);
      const int k0 = (c < 4) ? c * 32 : (c - 4) * 32;
      v16h a = load_a_frag16(rowp, k0, lane);
#pragma unroll
      for (int n0 = 0; n0 < 8; ++n0) {
        v16h b = load_b_frag(smem, 8, n0, c, lane);
        acc[n0] = __builtin_amdgcn_wmma_f32_16x16x32_f16(false, a, false, b,
                                                         (short)0, acc[n0],
                                                         false, false);
      }
    }

    float ss[8];
#pragma unroll
    for (int r = 0; r < 8; ++r) ss[r] = 0.f;
#pragma unroll
    for (int n0 = 0; n0 < 8; ++n0) {
#pragma unroll
      for (int r = 0; r < 8; ++r) {
        float x = lrelu(acc[n0][r] + hb[n0]);
        acc[n0][r] = x;
        ss[r] += x * x;
      }
    }
#pragma unroll
    for (int r = 0; r < 8; ++r) {
#pragma unroll
      for (int m = 1; m <= 8; m <<= 1) ss[r] += __shfl_xor(ss[r], m, 32);
    }
    float innorm[8];
#pragma unroll
    for (int r = 0; r < 8; ++r) innorm[r] = 1.f / (sqrtf(ss[r]) + 1e-6f);

    // one address per tile; all 64 stores use constant immediate offsets
    float* outp = out + (size_t)(base + hi) * HH + (lane & 15);
#pragma unroll
    for (int n0 = 0; n0 < 8; ++n0) {
#pragma unroll
      for (int r = 0; r < 8; ++r)
        outp[n0 * 16 + r * HH] =
            g[n0] * (acc[n0][r] * innorm[r] - mn[n0]) * iv[n0] + bt[n0];
    }
  }
}

// ---------------------------------------------------------------------------
extern "C" void kernel_launch(void* const* d_in, const int* in_sizes, int n_in,
                              void* d_out, int out_size, void* d_ws, size_t ws_size,
                              hipStream_t stream) {
  (void)in_sizes; (void)n_in; (void)out_size; (void)ws_size;
  const float* E     = (const float*)d_in[0];
  // d_in[1] ("weights") unused by call_att
  const int*   nbr   = (const int*)d_in[2];
  const float* Wq    = (const float*)d_in[3];
  const float* bq    = (const float*)d_in[4];
  const float* Wk    = (const float*)d_in[5];
  const float* bk    = (const float*)d_in[6];
  const float* Wv    = (const float*)d_in[7];
  const float* bv    = (const float*)d_in[8];
  const float* W1    = (const float*)d_in[9];
  const float* b1    = (const float*)d_in[10];
  const float* gamma = (const float*)d_in[11];
  const float* beta  = (const float*)d_in[12];
  const float* mean  = (const float*)d_in[13];
  const float* var   = (const float*)d_in[14];

  // workspace layout (all 16B aligned; FWk/FWv contiguous for 64KB staging)
  float*    Q   = (float*)d_ws;                        // NN*HH f32
  _Float16* E16 = (_Float16*)(Q + (size_t)NN * HH);    // NN*FF f16
  _Float16* P16 = E16 + (size_t)NN * FF;               // NN*HH f16
  _Float16* FWq = P16 + (size_t)NN * HH;               // FF*HH f16 frags
  _Float16* FWk = FWq + FF * HH;
  _Float16* FWv = FWk + FF * HH;
  _Float16* FW1 = FWv + FF * HH;                       // (FF+HH)*HH f16 frags
  float* out = (float*)d_out;

  hipLaunchKernelGGL(k_prep, dim3(1024), dim3(256), 0, stream,
                     E, Wq, Wk, Wv, W1, E16, FWq, FWk, FWv, FW1);
  hipLaunchKernelGGL(k_qproj, dim3(256), dim3(256),
                     FF * HH * sizeof(_Float16), stream, E16, FWq, bq, Q);
  hipLaunchKernelGGL(k_attn, dim3(512), dim3(256),
                     2 * FF * HH * sizeof(_Float16), stream,
                     E16, nbr, FWk, bk, bv, Q, P16);
  hipLaunchKernelGGL(k_out, dim3(256), dim3(256),
                     (FF + HH) * HH * sizeof(_Float16), stream,
                     E16, P16, FW1, b1, gamma, beta, mean, var, out);
}
